// VectorQuantizerEMA_1632087573352
// MI455X (gfx1250) — matrix operand order repaired and checked by reading, not compiled, not run on hardware
//
#include <hip/hip_runtime.h>
#include <hip/hip_bf16.h>

#define NUM_CODES 8192
#define EMBED_DIM 256
#define N_TOKENS 32768
#define BETA 0.25f
#define DECAY 0.99f
#define USAGE_LAMBDA 0.005f
#define CSPLIT 8
#define CODES_PER_SPLIT (NUM_CODES / CSPLIT)     // 1024
#define TILES_PER_SPLIT (CODES_PER_SPLIT / 16)   // 64
#define ROWS_PER_WG 64                           // 4 waves x 16 rows
#define LDS_STRIDE 264                           // halves; 528B rows -> lane nl at bank 4*nl

typedef __attribute__((ext_vector_type(16))) _Float16 v16h;
typedef __attribute__((ext_vector_type(8)))  _Float16 h8;
typedef __attribute__((ext_vector_type(8)))  float    v8f;

// ---------------------------------------------------- CDNA5 async LDS copy
// Copies 64B (4 x b128) global -> LDS without VGPR staging; tracked by
// ASYNCcnt. INST_OFFSET is added to BOTH the LDS and global addresses
// (CDNA5 ISA pseudocode), so one base pair serves all four.
__device__ __forceinline__ void async_copy_64B(uint32_t lds_off, const void* gptr) {
    asm volatile(
        "global_load_async_to_lds_b128 %0, %1, off\n\t"
        "global_load_async_to_lds_b128 %0, %1, off offset:16\n\t"
        "global_load_async_to_lds_b128 %0, %1, off offset:32\n\t"
        "global_load_async_to_lds_b128 %0, %1, off offset:48"
        :: "v"(lds_off), "v"(gptr) : "memory");
}
__device__ __forceinline__ void wait_async0() {
    asm volatile("s_wait_asynccnt 0x0" ::: "memory");
}

// ---------------------------------------------------------------- utilities
__global__ void zero_f32(float* __restrict__ p, int n) {
    int i = blockIdx.x * blockDim.x + threadIdx.x;
    if (i < n) p[i] = 0.0f;
}

// One wave per row: L2-normalize a fp32 row and emit f16.
__global__ __launch_bounds__(256)
void normalize_rows_f16(const float* __restrict__ src, _Float16* __restrict__ dst, int nrows) {
    int wave = threadIdx.x >> 5;
    int lane = threadIdx.x & 31;
    int row  = blockIdx.x * 8 + wave;
    if (row >= nrows) return;
    const float* r = src + (size_t)row * EMBED_DIM;
    float v[8];
    float ss = 0.0f;
    #pragma unroll
    for (int j = 0; j < 8; ++j) { v[j] = r[lane * 8 + j]; ss += v[j] * v[j]; }
    #pragma unroll
    for (int m = 1; m < 32; m <<= 1) ss += __shfl_xor(ss, m, 32);
    float s = 1.0f / fmaxf(sqrtf(ss), 1e-12f);
    _Float16* o = dst + (size_t)row * EMBED_DIM;
    #pragma unroll
    for (int j = 0; j < 8; ++j) o[lane * 8 + j] = (_Float16)(v[j] * s);
}

// ------------------------------------------------- WMMA similarity + argmax
// 4 waves per workgroup; each wave owns a 16-row A tile (register resident,
// full K=256). The 16-code B tile (8KB of f16 codebook, contiguous in memory)
// is double-buffered in LDS via async global->LDS copies issued BEFORE the
// WMMA sequence, so the L2 round-trip overlaps the matrix math.
__global__ __launch_bounds__(128)
void sim_argmax_wmma(const _Float16* __restrict__ x16, const _Float16* __restrict__ cb16,
                     float* __restrict__ pval, int* __restrict__ pidx) {
    __shared__ __align__(16) _Float16 ldsB[2][16 * LDS_STRIDE];

    const int tid  = threadIdx.x;
    const int wave = tid >> 5;
    const int lane = tid & 31;
    const int h    = lane >> 4;      // 16-lane half
    const int nl   = lane & 15;
    const int rowbase = blockIdx.x * ROWS_PER_WG + wave * 16;
    const int cbase0  = blockIdx.y * CODES_PER_SPLIT;

    // Preload A fragments: lane holds row m = rowbase + nl, K chunks per half.
    const _Float16* arow = x16 + (size_t)(rowbase + nl) * EMBED_DIM;
    v16h a[8];
    #pragma unroll
    for (int kb = 0; kb < 8; ++kb) {
        const int kk = kb * 32;
        h8 lo = *reinterpret_cast<const h8*>(arow + kk + h * 8);
        h8 hi = *reinterpret_cast<const h8*>(arow + kk + 16 + h * 8);
        #pragma unroll
        for (int j = 0; j < 8; ++j) { a[kb][j] = lo[j]; a[kb][8 + j] = hi[j]; }
    }

    float bestv[8];
    int   besti[8];
    #pragma unroll
    for (int r = 0; r < 8; ++r) { bestv[r] = -3.0e38f; besti[r] = 0x7fffffff; }

    // Cooperative copy mapping: row crow = tid/8, 8 threads x 64B per 512B row.
    const int crow = tid >> 3;
    const int csub = tid & 7;
    const uint32_t lds_dst0 =
        (uint32_t)(uintptr_t)&ldsB[0][crow * LDS_STRIDE + csub * 32];
    const uint32_t lds_dst1 =
        (uint32_t)(uintptr_t)&ldsB[1][crow * LDS_STRIDE + csub * 32];
    const _Float16* gsrc = cb16 + ((size_t)cbase0 + crow) * EMBED_DIM + csub * 32;

    // Prime the pipeline: tile 0 -> ldsB[0]
    async_copy_64B(lds_dst0, gsrc);
    wait_async0();
    __syncthreads();

    for (int t = 0; t < TILES_PER_SPLIT; ++t) {
        const int buf = t & 1;

        // Kick off the next tile's async copy FIRST (overlaps the WMMAs).
        if (t + 1 < TILES_PER_SPLIT) {
            async_copy_64B(buf ? lds_dst0 : lds_dst1,
                           gsrc + (size_t)(t + 1) * 16 * EMBED_DIM);
        }

        // 8 WMMAs over K=256 from LDS-resident B tile.
        const _Float16* brow = &ldsB[buf][nl * LDS_STRIDE];
        v8f c = {};
        #pragma unroll
        for (int kb = 0; kb < 8; ++kb) {
            const int kk = kb * 32;
            h8 blo = *reinterpret_cast<const h8*>(brow + kk + h * 8);
            h8 bhi = *reinterpret_cast<const h8*>(brow + kk + 16 + h * 8);
            v16h b;
            #pragma unroll
            for (int j = 0; j < 8; ++j) { b[j] = blo[j]; b[8 + j] = bhi[j]; }
            c = __builtin_amdgcn_wmma_f32_16x16x32_f16(false, a[kb], false, b,
                                                       (short)0, c, false, false);
        }

        const int code = cbase0 + t * 16 + nl;       // D column for this lane
        #pragma unroll
        for (int r = 0; r < 8; ++r) {
            float v = c[r];
            if (v > bestv[r] || (v == bestv[r] && code < besti[r])) {
                bestv[r] = v; besti[r] = code;
            }
        }

        // Next tile's copy must have landed before anyone reads it.
        wait_async0();
        __syncthreads();
    }

    // Reduce across the 16 lanes of each half (xor masks < 16 stay in-half).
    #pragma unroll
    for (int r = 0; r < 8; ++r) {
        #pragma unroll
        for (int mk = 1; mk < 16; mk <<= 1) {
            float ov = __shfl_xor(bestv[r], mk, 32);
            int   oi = __shfl_xor(besti[r], mk, 32);
            if (ov > bestv[r] || (ov == bestv[r] && oi < besti[r])) {
                bestv[r] = ov; besti[r] = oi;
            }
        }
    }
    if (nl == 0) {
        #pragma unroll
        for (int r = 0; r < 8; ++r) {
            int row = rowbase + h * 8 + r;           // D layout: half 1 -> rows 8..15
            pval[(size_t)row * CSPLIT + blockIdx.y] = bestv[r];
            pidx[(size_t)row * CSPLIT + blockIdx.y] = besti[r];
        }
    }
}

// ------------------------------------------------------------- finalization
__global__ __launch_bounds__(256)
void argmax_finalize(const float* __restrict__ pval, const int* __restrict__ pidx,
                     int* __restrict__ idx_i, float* __restrict__ out_idx,
                     float* __restrict__ counts) {
    int t = blockIdx.x * blockDim.x + threadIdx.x;
    if (t >= N_TOKENS) return;
    float bv = -3.0e38f; int bi = 0x7fffffff;
    #pragma unroll
    for (int s = 0; s < CSPLIT; ++s) {
        float v = pval[(size_t)t * CSPLIT + s];
        int   i = pidx[(size_t)t * CSPLIT + s];
        if (v > bv || (v == bv && i < bi)) { bv = v; bi = i; }
    }
    idx_i[t]   = bi;
    out_idx[t] = (float)bi;
    atomicAdd(&counts[bi], 1.0f);
}

// Gather quant rows, accumulate SSE for commit loss, scatter-add x into sum_vec.
__global__ __launch_bounds__(256)
void quant_scatter(const float* __restrict__ x, const float* __restrict__ cb,
                   const int* __restrict__ idx_i, float* __restrict__ quant,
                   float* __restrict__ sumvec, float* __restrict__ sse) {
    int wave = threadIdx.x >> 5, lane = threadIdx.x & 31;
    int t = blockIdx.x * 8 + wave;
    if (t >= N_TOKENS) return;
    int k = idx_i[t];
    const float* xr = x  + (size_t)t * EMBED_DIM;
    const float* cr = cb + (size_t)k * EMBED_DIM;
    float* qr = quant  + (size_t)t * EMBED_DIM;
    float* sr = sumvec + (size_t)k * EMBED_DIM;
    float acc = 0.0f;
    #pragma unroll
    for (int j = 0; j < 8; ++j) {
        int d = lane + j * 32;
        float xv = xr[d], cv = cr[d];
        qr[d] = cv;
        float df = xv - cv;
        acc += df * df;
        atomicAdd(&sr[d], xv);
    }
    #pragma unroll
    for (int m = 1; m < 32; m <<= 1) acc += __shfl_xor(acc, m, 32);
    if (lane == 0) atomicAdd(sse, acc);
}

// vq_loss / entropy / perplexity (one workgroup).
__global__ __launch_bounds__(1024)
void losses_kernel(const float* __restrict__ counts, const float* __restrict__ sse,
                   float* __restrict__ out_scalars) {
    __shared__ float red[1024];
    int tid = threadIdx.x;
    float tot = 0.0f;
    for (int i = tid; i < NUM_CODES; i += 1024) tot += counts[i];
    red[tid] = tot; __syncthreads();
    for (int s = 512; s > 0; s >>= 1) { if (tid < s) red[tid] += red[tid + s]; __syncthreads(); }
    float total = red[0];
    __syncthreads();
    float ent = 0.0f;
    for (int i = tid; i < NUM_CODES; i += 1024) {
        float p = counts[i] / (total + 1e-6f);
        ent -= p * logf(p + 1e-12f);
    }
    red[tid] = ent; __syncthreads();
    for (int s = 512; s > 0; s >>= 1) { if (tid < s) red[tid] += red[tid + s]; __syncthreads(); }
    if (tid == 0) {
        float entropy = red[0];
        float commit  = BETA * (sse[0] / (float)(N_TOKENS * EMBED_DIM));
        float logC    = logf((float)NUM_CODES + 1e-12f);
        out_scalars[0] = commit + USAGE_LAMBDA * (logC - entropy);  // vq_loss
        out_scalars[1] = entropy;
        out_scalars[2] = expf(entropy);                             // perplexity
    }
}

__global__ __launch_bounds__(256)
void count_update(const float* __restrict__ ema_count, const float* __restrict__ counts,
                  float* __restrict__ new_count) {
    int k = blockIdx.x * blockDim.x + threadIdx.x;
    if (k >= NUM_CODES) return;
    float nc = ema_count[k] * DECAY + (1.0f - DECAY) * counts[k];
    // dead-code reinit: with ema_count init = 1, nc >= 0.99 so never dead; the
    // JAX randn reinit is unreproducible here, so only the count clamp is applied.
    new_count[k] = (nc < 1e-3f) ? 1.0f : nc;
}

__global__ __launch_bounds__(256)
void ema_elem(const float* __restrict__ ema_sum, const float* __restrict__ sumvec,
              const float* __restrict__ ema_count, const float* __restrict__ counts,
              float* __restrict__ new_sum, float* __restrict__ new_cb) {
    int i = blockIdx.x * blockDim.x + threadIdx.x;
    if (i >= NUM_CODES * EMBED_DIM) return;
    int k = i >> 8;   // EMBED_DIM == 256
    float nc = ema_count[k] * DECAY + (1.0f - DECAY) * counts[k];  // pre-dead value (matches ref)
    float ns = ema_sum[i] * DECAY + (1.0f - DECAY) * sumvec[i];
    new_sum[i] = ns;
    new_cb[i]  = ns / (nc + 1e-5f);
}

// ------------------------------------------------------------------- launch
extern "C" void kernel_launch(void* const* d_in, const int* in_sizes, int n_in,
                              void* d_out, int out_size, void* d_ws, size_t ws_size,
                              hipStream_t stream) {
    const float* x         = (const float*)d_in[0];
    const float* codebook  = (const float*)d_in[1];
    const float* ema_count = (const float*)d_in[2];
    const float* ema_sum   = (const float*)d_in[3];
    float* out = (float*)d_out;

    // Output layout (return order): quant, idx, vq_loss, entropy, perplexity,
    // counts, new_codebook, new_count, new_sum
    const size_t off_quant    = 0;
    const size_t off_idx      = off_quant + (size_t)N_TOKENS * EMBED_DIM;
    const size_t off_vq       = off_idx + N_TOKENS;           // 3 scalars
    const size_t off_counts   = off_vq + 3;
    const size_t off_newcb    = off_counts + NUM_CODES;
    const size_t off_newcount = off_newcb + (size_t)NUM_CODES * EMBED_DIM;
    const size_t off_newsum   = off_newcount + NUM_CODES;

    // Workspace carve-up (~30.5 MB)
    char* w = (char*)d_ws;
    _Float16* x16  = (_Float16*)w;  w += (size_t)N_TOKENS * EMBED_DIM * sizeof(_Float16);
    _Float16* cb16 = (_Float16*)w;  w += (size_t)NUM_CODES * EMBED_DIM * sizeof(_Float16);
    float* pval    = (float*)w;     w += (size_t)N_TOKENS * CSPLIT * sizeof(float);
    int*   pidx    = (int*)w;       w += (size_t)N_TOKENS * CSPLIT * sizeof(int);
    int*   idx_i   = (int*)w;       w += (size_t)N_TOKENS * sizeof(int);
    float* sumvec  = (float*)w;     w += (size_t)NUM_CODES * EMBED_DIM * sizeof(float);
    float* sse     = (float*)w;     w += 256;

    // Zero accumulators (graph replays don't re-poison).
    zero_f32<<<(NUM_CODES * EMBED_DIM + 255) / 256, 256, 0, stream>>>(sumvec, NUM_CODES * EMBED_DIM);
    zero_f32<<<(NUM_CODES + 255) / 256, 256, 0, stream>>>(out + off_counts, NUM_CODES);
    zero_f32<<<1, 256, 0, stream>>>(sse, 1);

    normalize_rows_f16<<<N_TOKENS / 8, 256, 0, stream>>>(x, x16, N_TOKENS);
    normalize_rows_f16<<<NUM_CODES / 8, 256, 0, stream>>>(codebook, cb16, NUM_CODES);

    dim3 g(N_TOKENS / ROWS_PER_WG, CSPLIT);
    sim_argmax_wmma<<<g, 128, 0, stream>>>(x16, cb16, pval, pidx);

    argmax_finalize<<<N_TOKENS / 256, 256, 0, stream>>>(pval, pidx, idx_i,
                                                        out + off_idx, out + off_counts);
    quant_scatter<<<N_TOKENS / 8, 256, 0, stream>>>(x, codebook, idx_i,
                                                    out + off_quant, sumvec, sse);
    losses_kernel<<<1, 1024, 0, stream>>>(out + off_counts, sse, out + off_vq);
    count_update<<<NUM_CODES / 256, 256, 0, stream>>>(ema_count, out + off_counts,
                                                      out + off_newcount);
    ema_elem<<<(NUM_CODES * EMBED_DIM) / 256, 256, 0, stream>>>(ema_sum, sumvec, ema_count,
                                                                out + off_counts,
                                                                out + off_newsum,
                                                                out + off_newcb);
}
